// MultiheadAttention_86569360818684
// MI455X (gfx1250) — matrix operand order repaired and checked by reading, not compiled, not run on hardware
//
#include <hip/hip_runtime.h>

typedef _Float16 half_t;
typedef __attribute__((ext_vector_type(4)))  _Float16 v4h;
typedef __attribute__((ext_vector_type(8)))  _Float16 v8h;
typedef __attribute__((ext_vector_type(16))) _Float16 v16h;
typedef __attribute__((ext_vector_type(4)))  float    v4f;
typedef __attribute__((ext_vector_type(8)))  float    v8f;

#define HID 768
#define NH  12
#define HD  64
#define WELEM (HID*HID)          // 589824

union AB16 { v16h v; v8h h[2]; };

#define WMMA_F16(A, B, C) \
  __builtin_amdgcn_wmma_f32_16x16x32_f16(false, (A).v, false, (B).v, (short)0, (C), false, false)

// ---------------------------------------------------------------------------
// Kernel 0: convert the four weight matrices fp32 -> f16 (layout preserved,
// W[o][k] row-major). wh = [Wq | Wk | Wv | Wo], 4*589824 halfs.
// ---------------------------------------------------------------------------
__global__ __launch_bounds__(256) void cvt_weights_kernel(
    const float* __restrict__ Wq, const float* __restrict__ Wk,
    const float* __restrict__ Wv, const float* __restrict__ Wo,
    half_t* __restrict__ wh) {
  int idx = blockIdx.x * 256 + threadIdx.x;          // 0 .. 4*WELEM-1
  int m = idx / WELEM;
  int r = idx - m * WELEM;
  const float* src = (m == 0) ? Wq : (m == 1) ? Wk : (m == 2) ? Wv : Wo;
  wh[idx] = (half_t)src[r];
}

// ---------------------------------------------------------------------------
// Kernel 1: fused QKV projection + per-token attention.
// One block = 32 tokens (2 M-subtiles), 256 threads = 8 waves.
//   phase 1: stage x[32][768] -> f16 in LDS
//   phase 2: 2x2 register-blocked WMMA GEMM (1 ds + 1 global b128 per wmma).
//            Epilogue stores Q and V at natural columns; K columns are stored
//            TRANSPOSED (c -> g*64+d with d=c/12, g=c%12) so the attention
//            phase reads K contiguously in d.
//   phase 3: each wave does attention for 4 tokens with packed-f16 math
//            (v_pk_fma_f16 on v8h chunks), writes ctx f16 to global.
// LDS: qkv16 147456 B + xh 49152 B (aliased by softmax scratch) = 196608 B.
// ---------------------------------------------------------------------------
__global__ __launch_bounds__(256) void qkv_attn_kernel(
    const float*  __restrict__ x,
    const half_t* __restrict__ wqkv,     // [3][768][768] f16
    const float*  __restrict__ bq, const float* __restrict__ bk,
    const float*  __restrict__ bv,
    half_t* __restrict__ ctx)            // [N][768] f16 out
{
  extern __shared__ char lds[];
  half_t* qkv16   = (half_t*)lds;                       // [32][2304]
  half_t* xh      = (half_t*)(lds + 32 * 2304 * 2);     // [32][768]
  float*  scratch = (float*) (lds + 32 * 2304 * 2);     // aliases xh (phase 3)

  const int tid  = threadIdx.x;
  const int lane = tid & 31;
  const int wave = tid >> 5;
  const int row0 = blockIdx.x * 32;

  // ---- phase 1: stage x tile, converting fp32 -> f16 -----------------------
  {
    const v4f* xg = (const v4f*)(x + (size_t)row0 * HID); // 6144 chunks of 4
    v4h* xd = (v4h*)xh;
#pragma unroll
    for (int i = 0; i < 24; ++i) {
      int c = tid + i * 256;
      v4f f = xg[c];
      v4h h;
      h[0] = (half_t)f[0]; h[1] = (half_t)f[1];
      h[2] = (half_t)f[2]; h[3] = (half_t)f[3];
      xd[c] = h;
    }
  }
  __syncthreads();

  // ---- phase 2: 2x2 register-blocked WMMA GEMM for Q,K,V ------------------
  const int n  = lane & 15;    // output column within tile / A row
  const int hi = lane >> 4;    // lane half-group
  const half_t* a0p = xh + (size_t)n * HID;          // M-subtile 0
  const half_t* a1p = xh + (size_t)(16 + n) * HID;   // M-subtile 1

  for (int i = 0; i < 9; ++i) {
    const int p   = wave + 8 * i;          // column-pair id, 0..71
    const int mat = p / 24;                // 0=Q 1=K 2=V (24 pairs per matrix)
    const int oc  = (p - mat * 24) * 32;   // column base within 768
    const half_t* wb0 = wqkv + (size_t)mat * WELEM + (size_t)(oc + n) * HID;
    const half_t* wb1 = wb0 + (size_t)16 * HID;

    // warm caches for the next column pair's weight rows
    if (i < 8) {
      const int pn   = wave + 8 * (i + 1);
      const int matn = pn / 24;
      const int ocn  = (pn - matn * 24) * 32;
      const half_t* nb = wqkv + (size_t)matn * WELEM + (size_t)(ocn + n) * HID;
      __builtin_prefetch(nb, 0, 1);
      __builtin_prefetch(nb + (size_t)16 * HID, 0, 1);
    }

    v8f acc00 = {}, acc01 = {}, acc10 = {}, acc11 = {};
#pragma unroll 4
    for (int kb = 0; kb < HID; kb += 32) {
      AB16 b0, b1, a0, a1;
      b0.h[0] = *(const v8h*)(wb0 + kb + hi * 8);
      b0.h[1] = *(const v8h*)(wb0 + kb + 16 + hi * 8);
      b1.h[0] = *(const v8h*)(wb1 + kb + hi * 8);
      b1.h[1] = *(const v8h*)(wb1 + kb + 16 + hi * 8);
      a0.h[0] = *(const v8h*)(a0p + kb + hi * 8);
      a0.h[1] = *(const v8h*)(a0p + kb + 16 + hi * 8);
      a1.h[0] = *(const v8h*)(a1p + kb + hi * 8);
      a1.h[1] = *(const v8h*)(a1p + kb + 16 + hi * 8);
      acc00 = WMMA_F16(a0, b0, acc00);
      acc01 = WMMA_F16(a0, b1, acc01);
      acc10 = WMMA_F16(a1, b0, acc10);
      acc11 = WMMA_F16(a1, b1, acc11);
    }

    const float* bias = (mat == 0) ? bq : (mat == 1) ? bk : bv;
    const int c0 = oc + n;                 // original output column (0..767)
    const int c1 = c0 + 16;
    const float bv0 = bias[c0];
    const float bv1 = bias[c1];
    // LDS column: Q/V natural; K transposed (Kh[d][g]=K[d*12+g] -> Kt[g][d])
    int col0, col1;
    if (mat == 1) {
      col0 = HID + (c0 - (c0 / NH) * NH) * HD + (c0 / NH);
      col1 = HID + (c1 - (c1 / NH) * NH) * HD + (c1 / NH);
    } else {
      col0 = mat * HID + c0;
      col1 = mat * HID + c1;
    }
#pragma unroll
    for (int r = 0; r < 8; ++r) {
      const int rr = r + 8 * hi;
      qkv16[ rr       * 2304 + col0] = (half_t)(acc00[r] + bv0);
      qkv16[ rr       * 2304 + col1] = (half_t)(acc01[r] + bv1);
      qkv16[(rr + 16) * 2304 + col0] = (half_t)(acc10[r] + bv0);
      qkv16[(rr + 16) * 2304 + col1] = (half_t)(acc11[r] + bv1);
    }
  }
  __syncthreads();

  // ---- phase 3: per-token attention (wave handles 4 tokens) ---------------
  float* sc = scratch + wave * 144;        // 12x12 scores / attn for this wave
  for (int tt = 0; tt < 4; ++tt) {
    const int tok = wave * 4 + tt;
    const half_t* Q  = qkv16 + tok * 2304;
    const half_t* Kt = Q + HID;            // transposed: Kt[g*64 + d]
    const half_t* V  = Q + 2 * HID;

    // scores[h][g] = 0.125 * sum_d Q[h*64+d] * Kt[g*64+d]  (packed f16)
    for (int idx = lane; idx < 144; idx += 32) {
      const int h = idx / NH, g = idx - h * NH;
      const v8h* qv = (const v8h*)(Q + h * HD);
      const v8h* kv = (const v8h*)(Kt + g * HD);
      v8h pacc = {};
#pragma unroll
      for (int c = 0; c < 8; ++c) pacc += qv[c] * kv[c];
      float s = 0.f;
#pragma unroll
      for (int j = 0; j < 8; ++j) s += (float)pacc[j];
      sc[idx] = s * 0.125f;
    }
    __syncthreads();

    // softmax over g: lane h owns row h
    if (lane < NH) {
      float* row = sc + lane * NH;
      float mx = row[0];
#pragma unroll
      for (int g = 1; g < NH; ++g) mx = fmaxf(mx, row[g]);
      float e[NH], sum = 0.f;
#pragma unroll
      for (int g = 0; g < NH; ++g) { e[g] = __expf(row[g] - mx); sum += e[g]; }
      const float inv = 1.f / sum;
#pragma unroll
      for (int g = 0; g < NH; ++g) row[g] = e[g] * inv;
    }
    __syncthreads();

    // ctx[h*64+d] = sum_g attn[h][g] * V[g*64+d]; lane owns 8-wide d chunks
    half_t* outp = ctx + (size_t)(row0 + tok) * HID;
    for (int cc = lane; cc < 96; cc += 32) {
      const int h  = cc >> 3;              // chunk's head
      const int dc = (cc & 7) * 8;         // d base within head
      const float* ah = sc + h * NH;
      v8h acc = {};
#pragma unroll
      for (int g = 0; g < NH; ++g) {
        const half_t w = (half_t)ah[g];
        const v8h wv = { w, w, w, w, w, w, w, w };
        acc += wv * *(const v8h*)(V + g * HD + dc);
      }
      *(v8h*)(outp + cc * 8) = acc;
    }
    __syncthreads();
  }
}

// ---------------------------------------------------------------------------
// Kernel 2: out = ctx @ Wo.T + bo   (fp32 output)
// One block = 64 tokens (4 M-subtiles), 8 waves x 6 column tiles; each B
// fragment feeds 4 WMMAs (0.5 global b128 per wmma).
// ---------------------------------------------------------------------------
__global__ __launch_bounds__(256) void out_proj_kernel(
    const half_t* __restrict__ ctx, const half_t* __restrict__ wo,
    const float* __restrict__ bo, float* __restrict__ out)
{
  extern __shared__ char lds2[];
  half_t* ch = (half_t*)lds2;              // [64][768]

  const int tid  = threadIdx.x;
  const int lane = tid & 31;
  const int wave = tid >> 5;
  const int row0 = blockIdx.x * 64;

  {
    const v8h* src = (const v8h*)(ctx + (size_t)row0 * HID); // 6144 chunks
    v8h* dst = (v8h*)ch;
#pragma unroll
    for (int i = 0; i < 24; ++i) dst[tid + i * 256] = src[tid + i * 256];
  }
  __syncthreads();

  const int n  = lane & 15;
  const int hi = lane >> 4;
  const half_t* a0p = ch + (size_t)n * HID;
  const half_t* a1p = a0p + (size_t)16 * HID;
  const half_t* a2p = a0p + (size_t)32 * HID;
  const half_t* a3p = a0p + (size_t)48 * HID;

  for (int i = 0; i < 6; ++i) {
    const int t  = wave + 8 * i;           // 0..47
    const int oc = t * 16;
    const half_t* wb = wo + (size_t)(oc + n) * HID;
    if (i < 5) {
      __builtin_prefetch(wo + (size_t)(oc + 128 + n) * HID, 0, 1);
    }

    v8f acc0 = {}, acc1 = {}, acc2 = {}, acc3 = {};
#pragma unroll 4
    for (int kb = 0; kb < HID; kb += 32) {
      AB16 b, a0, a1, a2, a3;
      b.h[0]  = *(const v8h*)(wb + kb + hi * 8);
      b.h[1]  = *(const v8h*)(wb + kb + 16 + hi * 8);
      a0.h[0] = *(const v8h*)(a0p + kb + hi * 8);
      a0.h[1] = *(const v8h*)(a0p + kb + 16 + hi * 8);
      a1.h[0] = *(const v8h*)(a1p + kb + hi * 8);
      a1.h[1] = *(const v8h*)(a1p + kb + 16 + hi * 8);
      a2.h[0] = *(const v8h*)(a2p + kb + hi * 8);
      a2.h[1] = *(const v8h*)(a2p + kb + 16 + hi * 8);
      a3.h[0] = *(const v8h*)(a3p + kb + hi * 8);
      a3.h[1] = *(const v8h*)(a3p + kb + 16 + hi * 8);
      acc0 = WMMA_F16(a0, b, acc0);
      acc1 = WMMA_F16(a1, b, acc1);
      acc2 = WMMA_F16(a2, b, acc2);
      acc3 = WMMA_F16(a3, b, acc3);
    }

    const float bval = bo[oc + n];
    float* o0 = out + (size_t)(row0 + 8 * hi) * HID + oc + n;
#pragma unroll
    for (int r = 0; r < 8; ++r) {
      o0[(size_t)(r     ) * HID] = acc0[r] + bval;
      o0[(size_t)(r + 16) * HID] = acc1[r] + bval;
      o0[(size_t)(r + 32) * HID] = acc2[r] + bval;
      o0[(size_t)(r + 48) * HID] = acc3[r] + bval;
    }
  }
}

// ---------------------------------------------------------------------------
extern "C" void kernel_launch(void* const* d_in, const int* in_sizes, int n_in,
                              void* d_out, int out_size, void* d_ws, size_t ws_size,
                              hipStream_t stream) {
  const float* x  = (const float*)d_in[0];
  const float* Wq = (const float*)d_in[1];
  const float* bq = (const float*)d_in[2];
  const float* Wk = (const float*)d_in[3];
  const float* bk = (const float*)d_in[4];
  const float* Wv = (const float*)d_in[5];
  const float* bv = (const float*)d_in[6];
  const float* Wo = (const float*)d_in[7];
  const float* bo = (const float*)d_in[8];
  float* out = (float*)d_out;

  const int n_tok = in_sizes[0] / HID;     // 65536

  // workspace layout (halfs): [Wq|Wk|Wv]_h, Wo_h, ctx_h  (~105 MB total)
  half_t* wh   = (half_t*)d_ws;
  half_t* wqkv = wh;                        // 3*WELEM halfs
  half_t* wo_h = wh + (size_t)3 * WELEM;    // WELEM halfs
  half_t* ctxh = wh + (size_t)4 * WELEM;    // n_tok*HID halfs

  // 1) convert weights to f16
  cvt_weights_kernel<<<(4 * WELEM) / 256, 256, 0, stream>>>(Wq, Wk, Wv, Wo, wh);

  // 2) fused QKV + attention (32 tokens per block, 196608 B LDS)
  const size_t ldsB = (size_t)32 * 2304 * 2 + (size_t)32 * HID * 2;
  qkv_attn_kernel<<<n_tok / 32, 256, ldsB, stream>>>(x, wqkv, bq, bk, bv, ctxh);

  // 3) output projection (64 tokens per block, 98304 B LDS)
  const size_t ldsC = (size_t)64 * HID * 2;
  out_proj_kernel<<<n_tok / 64, 256, ldsC, stream>>>(ctxh, wo_h, bo, out);
}
